// Attention_120259084784
// MI455X (gfx1250) — matrix operand order repaired and checked by reading, not compiled, not run on hardware
//
#include <hip/hip_runtime.h>
#include <hip/hip_bf16.h>

typedef __attribute__((ext_vector_type(16))) __bf16       v16bf;
typedef __attribute__((ext_vector_type(8)))  float        v8f;
typedef __attribute__((ext_vector_type(8)))  unsigned int v8u;   // 32-byte copy unit

__device__ __forceinline__ unsigned short f2bf(float f) {
    unsigned int u = __float_as_uint(f);
    unsigned int r = u + 0x7fffu + ((u >> 16) & 1u);   // round-to-nearest-even
    return (unsigned short)(r >> 16);
}

__device__ __forceinline__ v8f wmma_bf16(v16bf a, v16bf b, v8f c) {
    return __builtin_amdgcn_wmma_f32_16x16x32_bf16(false, a, false, b,
                                                   (short)0, c, false, false);
}

// A-fragment (16x32 bf16, MxK): lane holds row M=lane%16; 16 contiguous K at 16*half.
__device__ __forceinline__ v16bf load_a_frag(const unsigned short* rowp, int hlf) {
    union { v16bf v; uint4 q[2]; } u;
    u.q[0] = *(const uint4*)(rowp + 16 * hlf);
    u.q[1] = *(const uint4*)(rowp + 16 * hlf + 8);
    return u.v;
}

// B-fragment (32x16 bf16, KxN): lane holds col N=lane%16; 16 contiguous K at 16*half.
__device__ __forceinline__ v16bf load_b_frag(const unsigned short* colp, int hlf) {
    union { v16bf v; v8u u; } c;
    c.u = *(const v8u*)(colp + 16 * hlf);
    return c.v;
}

// LDS byte offset of a __shared__ object: generic LDS pointers carry the DS
// offset in addr[31:0] (ISA flat-aperture mapping).
__device__ __forceinline__ unsigned lds_off(const void* p) {
    return (unsigned)(uintptr_t)p;
}

// 64-byte global -> LDS async copy (4x b128); IOFFSET applies to both addresses.
__device__ __forceinline__ void async_cp64(unsigned dst, const void* src) {
    asm volatile("global_load_async_to_lds_b128 %0, %1, off"           :: "v"(dst), "v"(src) : "memory");
    asm volatile("global_load_async_to_lds_b128 %0, %1, off offset:16" :: "v"(dst), "v"(src) : "memory");
    asm volatile("global_load_async_to_lds_b128 %0, %1, off offset:32" :: "v"(dst), "v"(src) : "memory");
    asm volatile("global_load_async_to_lds_b128 %0, %1, off offset:48" :: "v"(dst), "v"(src) : "memory");
}

__device__ __forceinline__ void wait_async0() {
    asm volatile("s_wait_asynccnt 0" ::: "memory");
}

// ---------------- fp32 -> bf16 convert ----------------
__global__ void cvt4_kernel(const float4* __restrict__ in, uint2* __restrict__ out, int n4) {
    int i = blockIdx.x * blockDim.x + threadIdx.x;
    if (i >= n4) return;
    float4 v = in[i];
    uint2 p;
    p.x = (unsigned)f2bf(v.x) | ((unsigned)f2bf(v.y) << 16);
    p.y = (unsigned)f2bf(v.z) | ((unsigned)f2bf(v.w) << 16);
    out[i] = p;
}

// =====================================================================
// Shared tiled-GEMM body: 128x128 tile, 8 waves x (32x64), K-step 64,
// double-buffered LDS filled by async-to-LDS, one barrier per K-step.
// Accumulators acc[2][4]; A,B are K-contiguous (row-major, op = A * B^T).
// =====================================================================
#define GEMM_BODY(Aptr, Bptr, KDIM)                                              \
    __shared__ alignas(32) unsigned short At[2][128 * 64];                       \
    __shared__ alignas(32) unsigned short Bt[2][128 * 64];                       \
    const int tid = threadIdx.x;                                                 \
    const int lane = tid & 31, wv = tid >> 5;                                    \
    const int lr = lane & 15, hlf = lane >> 4;                                   \
    const int wm = (wv & 3) * 32;                                                \
    const int wn = (wv >> 2) * 64;                                               \
    const int mBase = blockIdx.y * 128;                                          \
    const int nBase = blockIdx.x * 128;                                          \
    v8f acc[2][4] = {};                                                          \
    const int lrow = tid >> 1;                                                   \
    const int lcg  = (tid & 1) * 32;                                             \
    const unsigned short* Arow = (Aptr) + (size_t)(mBase + lrow) * (KDIM) + lcg; \
    const unsigned short* Brow = (Bptr) + (size_t)(nBase + lrow) * (KDIM) + lcg; \
    const unsigned atw[2] = { lds_off(&At[0][lrow * 64 + lcg]),                  \
                              lds_off(&At[1][lrow * 64 + lcg]) };                \
    const unsigned btw[2] = { lds_off(&Bt[0][lrow * 64 + lcg]),                  \
                              lds_off(&Bt[1][lrow * 64 + lcg]) };                \
    async_cp64(atw[0], Arow);                                                    \
    async_cp64(btw[0], Brow);                                                    \
    int buf = 0;                                                                 \
    for (int k0 = 0; k0 < (KDIM); k0 += 64, buf ^= 1) {                          \
        wait_async0();                                                           \
        __syncthreads();                                                         \
        if (k0 + 64 < (KDIM)) {            /* refill other buffer */             \
            async_cp64(atw[buf ^ 1], Arow + k0 + 64);                            \
            async_cp64(btw[buf ^ 1], Brow + k0 + 64);                            \
        }                                                                        \
        _Pragma("unroll")                                                        \
        for (int kk = 0; kk < 64; kk += 32) {                                    \
            v16bf a[2], b[4];                                                    \
            _Pragma("unroll")                                                    \
            for (int mi = 0; mi < 2; ++mi)                                       \
                a[mi] = load_a_frag(&At[buf][(wm + mi * 16 + lr) * 64 + kk], hlf);\
            _Pragma("unroll")                                                    \
            for (int ni = 0; ni < 4; ++ni)                                       \
                b[ni] = load_b_frag(&Bt[buf][(wn + ni * 16 + lr) * 64 + kk], hlf);\
            _Pragma("unroll")                                                    \
            for (int mi = 0; mi < 2; ++mi)                                       \
                _Pragma("unroll")                                                \
                for (int ni = 0; ni < 4; ++ni)                                   \
                    acc[mi][ni] = wmma_bf16(a[mi], b[ni], acc[mi][ni]);          \
        }                                                                        \
    }

// ---- QKV GEMM: [8192,768] x [2304,768]^T -> qkv[3][96][1024][64], Q scaled ----
__global__ __launch_bounds__(256) void qkv_gemm(const unsigned short* __restrict__ X,
                                                const unsigned short* __restrict__ W,
                                                unsigned short* __restrict__ qkv) {
    GEMM_BODY(X, W, 768)

    // 768 % 128 == 0 -> s and scale are uniform across the workgroup
    const int s = nBase / 768;
    const int remBase = nBase - s * 768;
    const float scl = (s == 0) ? 0.125f : 1.0f;   // fold softmax scale into Q
#pragma unroll
    for (int mi = 0; mi < 2; ++mi) {
        int mrow = mBase + wm + mi * 16 + 8 * hlf;
#pragma unroll
        for (int ni = 0; ni < 4; ++ni) {
            int rem = remBase + wn + ni * 16 + lr;
            int h = rem >> 6, d = rem & 63;
#pragma unroll
            for (int r = 0; r < 8; ++r) {
                int m  = mrow + r;
                int bb = m >> 10, nn = m & 1023;
                size_t idx = ((((size_t)s * 96) + (size_t)bb * 12 + h) * 1024 + nn) * 64 + d;
                qkv[idx] = f2bf(acc[mi][ni][r] * scl);
            }
        }
    }
}

// ---- Proj GEMM: [8192,768] x [768,768]^T + bias -> f32 out ----
__global__ __launch_bounds__(256) void proj_gemm(const unsigned short* __restrict__ A0,
                                                 const unsigned short* __restrict__ W,
                                                 const float* __restrict__ bias,
                                                 float* __restrict__ out) {
    GEMM_BODY(A0, W, 768)

#pragma unroll
    for (int mi = 0; mi < 2; ++mi) {
        int mrow = mBase + wm + mi * 16 + 8 * hlf;
#pragma unroll
        for (int ni = 0; ni < 4; ++ni) {
            int j = nBase + wn + ni * 16 + lr;
            float bz = bias[j];
#pragma unroll
            for (int r = 0; r < 8; ++r)
                out[(size_t)(mrow + r) * 768 + j] = acc[mi][ni][r] + bz;
        }
    }
}

// ---------------- Flash attention: grid (8 qblocks, 96 heads), 256 thr ----------
__global__ __launch_bounds__(256) void attn_kernel(const unsigned short* __restrict__ qkv,
                                                   unsigned short* __restrict__ aout) {
    constexpr int N = 1024, HD = 64;
    __shared__ alignas(32) unsigned short Kt[128 * 64];   // [key][d]
    __shared__ alignas(32) unsigned short VT[64 * 128];   // [d][key]
    __shared__ alignas(32) unsigned short Pst[8 * 16 * 32];

    const int tid = threadIdx.x, lane = tid & 31, wv = tid >> 5;
    const int lr = lane & 15, hlf = lane >> 4;
    const int bh = blockIdx.y;                                  // b*12+h
    const size_t headOff = (size_t)bh * N * HD;
    const unsigned short* qg = qkv + headOff;
    const unsigned short* kg = qkv + (size_t)96 * N * HD + headOff;
    const unsigned short* vg = qkv + (size_t)2 * 96 * N * HD + headOff;
    const int qbase = blockIdx.x * 128 + wv * 16;

    v16bf aQ[2];
#pragma unroll
    for (int c = 0; c < 2; ++c)
        aQ[c] = load_a_frag(qg + (size_t)(qbase + lr) * HD + c * 32, hlf);

    v8f o[4] = {};
    float mstat[8], lstat[8];
#pragma unroll
    for (int e = 0; e < 8; ++e) { mstat[e] = -3.0e38f; lstat[e] = 0.f; }

    const int krow = tid >> 1, kcg = (tid & 1) * 32;
    const unsigned ktw = lds_off(&Kt[krow * 64 + kcg]);

    for (int kb = 0; kb < 8; ++kb) {
        const int kbase = kb * 128;
        __syncthreads();                     // all waves done reading Kt/VT
        // K tile via async-to-LDS (64B per thread)
        async_cp64(ktw, kg + (size_t)(kbase + krow) * HD + kcg);
        // V tile, transposed through VGPRs
#pragma unroll
        for (int i = 0; i < 8; ++i) {
            int e4  = (i * 256 + tid) * 4;
            int key = e4 >> 6, d = e4 & 63;
            uint2 pk = *(const uint2*)(vg + (size_t)(kbase + key) * HD + d);
            VT[(d + 0) * 128 + key] = (unsigned short)(pk.x & 0xffff);
            VT[(d + 1) * 128 + key] = (unsigned short)(pk.x >> 16);
            VT[(d + 2) * 128 + key] = (unsigned short)(pk.y & 0xffff);
            VT[(d + 3) * 128 + key] = (unsigned short)(pk.y >> 16);
        }
        wait_async0();
        __syncthreads();

        for (int c4 = 0; c4 < 4; ++c4) {
            const int koff = c4 * 32;
            v8f s0 = {}, s1 = {};
            s0 = wmma_bf16(aQ[0], load_b_frag(Kt + (koff      + lr) * 64,      hlf), s0);
            s0 = wmma_bf16(aQ[1], load_b_frag(Kt + (koff      + lr) * 64 + 32, hlf), s0);
            s1 = wmma_bf16(aQ[0], load_b_frag(Kt + (koff + 16 + lr) * 64,      hlf), s1);
            s1 = wmma_bf16(aQ[1], load_b_frag(Kt + (koff + 16 + lr) * 64 + 32, hlf), s1);

            // online softmax (elem e -> row e+8*half; 16 lanes per row)
#pragma unroll
            for (int e = 0; e < 8; ++e) {
                float cm = fmaxf(s0[e], s1[e]);
                cm = fmaxf(cm, __shfl_xor(cm, 1, 32));
                cm = fmaxf(cm, __shfl_xor(cm, 2, 32));
                cm = fmaxf(cm, __shfl_xor(cm, 4, 32));
                cm = fmaxf(cm, __shfl_xor(cm, 8, 32));
                float mn = fmaxf(mstat[e], cm);
                float al = __expf(mstat[e] - mn);
                float p0 = __expf(s0[e] - mn);
                float p1 = __expf(s1[e] - mn);
                float rs = p0 + p1;
                rs += __shfl_xor(rs, 1, 32);
                rs += __shfl_xor(rs, 2, 32);
                rs += __shfl_xor(rs, 4, 32);
                rs += __shfl_xor(rs, 8, 32);
                lstat[e] = lstat[e] * al + rs;
                mstat[e] = mn;
                s0[e] = p0; s1[e] = p1;
                o[0][e] *= al; o[1][e] *= al; o[2][e] *= al; o[3][e] *= al;
            }

            // C-layout -> A-layout via per-wave LDS pad (same-wave DS ops in order)
            unsigned short* pw = Pst + wv * 512;
#pragma unroll
            for (int e = 0; e < 8; ++e) {
                int row = e + 8 * hlf;
                pw[row * 32 + lr]      = f2bf(s0[e]);
                pw[row * 32 + 16 + lr] = f2bf(s1[e]);
            }
            v16bf aP = load_a_frag(pw + lr * 32, hlf);
#pragma unroll
            for (int ni = 0; ni < 4; ++ni) {
                v16bf bV = load_b_frag(VT + (ni * 16 + lr) * 128 + koff, hlf);
                o[ni] = wmma_bf16(aP, bV, o[ni]);
            }
        }
    }

    // finalize: aout[b][n][h*64+d] (bf16 input to proj GEMM)
    const int b = bh / 12, h = bh - b * 12;
#pragma unroll
    for (int e = 0; e < 8; ++e) {
        float inv = 1.0f / lstat[e];
        int nq = qbase + e + 8 * hlf;
        size_t base = ((size_t)b * 1024 + nq) * 768 + h * 64;
#pragma unroll
        for (int ni = 0; ni < 4; ++ni)
            aout[base + ni * 16 + lr] = f2bf(o[ni][e] * inv);
    }
}

extern "C" void kernel_launch(void* const* d_in, const int* in_sizes, int n_in,
                              void* d_out, int out_size, void* d_ws, size_t ws_size,
                              hipStream_t stream) {
    const float* x      = (const float*)d_in[0];   // [8,1024,768]
    const float* w_qkv  = (const float*)d_in[1];   // [2304,768]
    const float* w_proj = (const float*)d_in[2];   // [768,768]
    const float* b_proj = (const float*)d_in[3];   // [768]
    float* out = (float*)d_out;

    unsigned short* xb    = (unsigned short*)d_ws;             // 8192*768
    unsigned short* wqb   = xb  + (size_t)8192 * 768;          // 2304*768
    unsigned short* wpb   = wqb + (size_t)2304 * 768;          // 768*768
    unsigned short* qkvb  = wpb + (size_t)768 * 768;           // 3*96*1024*64
    unsigned short* attnb = qkvb + (size_t)3 * 96 * 1024 * 64; // 8192*768

    {   // fp32 -> bf16 converts
        int n4 = (8192 * 768) / 4;
        cvt4_kernel<<<(n4 + 255) / 256, 256, 0, stream>>>((const float4*)x, (uint2*)xb, n4);
        n4 = (2304 * 768) / 4;
        cvt4_kernel<<<(n4 + 255) / 256, 256, 0, stream>>>((const float4*)w_qkv, (uint2*)wqb, n4);
        n4 = (768 * 768) / 4;
        cvt4_kernel<<<(n4 + 255) / 256, 256, 0, stream>>>((const float4*)w_proj, (uint2*)wpb, n4);
    }

    qkv_gemm<<<dim3(2304 / 128, 8192 / 128), 256, 0, stream>>>(xb, wqb, qkvb);
    attn_kernel<<<dim3(1024 / 128, 96), 256, 0, stream>>>(qkvb, attnb);
    proj_gemm<<<dim3(768 / 128, 8192 / 128), 256, 0, stream>>>(attnb, wpb, b_proj, out);
}
